// WindowAttention_51110110822667
// MI455X (gfx1250) — matrix operand order repaired and checked by reading, not compiled, not run on hardware
//
#include <hip/hip_runtime.h>
#include <hip/hip_bf16.h>

// ---------------------------------------------------------------------------
// Fused Swin window attention for gfx1250 (MI455X), wave32 + bf16 WMMA.
//   x:(4096,49,192) -> qkv -> per-head attn (bias+mask+softmax) -> proj -> out
// One workgroup (256 threads = 8 wave32) per window; everything except the
// final output stays in LDS. Matmuls use v_wmma_f32_16x16x32_bf16; the window
// mask is staged to LDS with global_load_async_to_lds_b32 (ASYNCcnt).
// ---------------------------------------------------------------------------

#define DIM     192
#define HEADS   6
#define HD      32
#define NTOK    49
#define NP      64          // padded token count (4 x 16 M-tiles)
#define NWIN    4096
#define NMASK   64
#define QKV_N   576         // 3*DIM

// LDS strides (elements). Multiples of 16 keep 32B vector accesses aligned.
#define XS_STRIDE   208     // 64 x 208 bf16  (x tile, later reused for attn-out)
#define QKV_STRIDE  400     // 64 x 400 bf16  (q|k blocks only, 384 cols + pad)
#define VT_STRIDE   80      // 192 x 80 bf16  (V transposed: vT[d][token])
#define SC_STRIDE   68      // 64 x 68  f32   (scores, one head at a time)
#define AT_STRIDE   80      // 64 x 80  bf16  (softmaxed probs)

#define XS_BYTES   (64 * XS_STRIDE * 2)             // 26624
#define QKV_BYTES  (64 * QKV_STRIDE * 2)            // 51200
#define VT_BYTES   (DIM * VT_STRIDE * 2)            // 30720
#define SC_BYTES   (64 * SC_STRIDE * 4)             // 17408
#define AT_BYTES   (64 * AT_STRIDE * 2)             // 10240
#define MK_BYTES   (2432 * 4)                       // 9728 (49*49 mask + pad)
#define SMEM_BYTES (XS_BYTES + QKV_BYTES + VT_BYTES + SC_BYTES + AT_BYTES + MK_BYTES)

typedef __attribute__((ext_vector_type(16))) __bf16 v16bf;
typedef __attribute__((ext_vector_type(8)))  __bf16 v8bf;
typedef __attribute__((ext_vector_type(8)))  float  v8f;

// ---- WMMA fragment helpers (A and B both use contiguous K order, k = 16*hi + s) ------

// A-matrix 16x32 bf16: lanes 0-15 hold row M=lane, K 0..15;
// lanes 16-31 hold the same rows, K 16..31.
__device__ __forceinline__ v16bf load_a16x32(const __bf16* base, int stride,
                                             int m0, int k0, int lane) {
  const int l  = lane & 15;
  const int hi = (lane >> 4) & 1;
  return *(const v16bf*)(base + (m0 + l) * stride + k0 + hi * 16);
}

// B-matrix 32x16 bf16 where memory holds Bt[n][k] row-major (k contiguous):
// lane = column n; lanes 0-15 K=0..15, lanes 16-31 K=16..31 -> contiguous load.
__device__ __forceinline__ v16bf load_b_kcontig(const __bf16* base, int stride,
                                                int n0, int k0, int lane) {
  const int l  = lane & 15;
  const int hi = (lane >> 4) & 1;
  return *(const v16bf*)(base + (n0 + l) * stride + k0 + hi * 16);
}

// B fragment built from f32 weights in global memory, layout w[n][k] row-major
// (so the needed 16 K-values per lane are one contiguous 64B read, L2-resident).
__device__ __forceinline__ v16bf load_b_gweight(const float* w, int ld,
                                                int n0, int k0, int lane) {
  const int l  = lane & 15;
  const int hi = (lane >> 4) & 1;
  const float* p = w + (size_t)(n0 + l) * ld + k0 + hi * 16;
  v16bf b;
#pragma unroll
  for (int i = 0; i < 16; ++i) b[i] = (__bf16)p[i];
  return b;
}

// C/D 16x16 f32: VGPR r holds M = m0 + r + 8*hi, N = n0 + lane&15.
__device__ __forceinline__ void store_d_bf16(__bf16* base, int stride, int m0,
                                             int n0, int lane, v8f c, float mul) {
  const int l  = lane & 15;
  const int hi = (lane >> 4) & 1;
#pragma unroll
  for (int r = 0; r < 8; ++r)
    base[(m0 + hi * 8 + r) * stride + n0 + l] = (__bf16)(c[r] * mul);
}

// Transposed store for the V block: D element (M=token j, N=head-dim d) goes to
// vT[d][j]; the 8 per-lane values are consecutive in j -> one 16B v8bf store.
__device__ __forceinline__ void store_d_vT(__bf16* vT, int m0, int d0,
                                           int lane, v8f c) {
  const int l  = lane & 15;
  const int hi = (lane >> 4) & 1;
  v8bf pk;
#pragma unroll
  for (int r = 0; r < 8; ++r) pk[r] = (__bf16)c[r];
  *(v8bf*)(vT + (d0 + l) * VT_STRIDE + m0 + hi * 8) = pk;
}

__device__ __forceinline__ void store_d_f32(float* base, int stride, int m0,
                                            int n0, int lane, v8f c) {
  const int l  = lane & 15;
  const int hi = (lane >> 4) & 1;
#pragma unroll
  for (int r = 0; r < 8; ++r)
    base[(m0 + hi * 8 + r) * stride + n0 + l] = c[r];
}

#define WMMA_BF16(A, B, C) \
  __builtin_amdgcn_wmma_f32_16x16x32_bf16(false, (A), false, (B), (short)0, (C), false, false)

// ---------------------------------------------------------------------------

__global__ void __launch_bounds__(256, 1)
win_attn_fused_kernel(const float* __restrict__ x,
                      const float* __restrict__ mask,
                      const float* __restrict__ qkv_w,
                      const float* __restrict__ qkv_b,
                      const float* __restrict__ proj_w,
                      const float* __restrict__ proj_b,
                      const float* __restrict__ bias_table,
                      const int*   __restrict__ rel_idx,
                      float*       __restrict__ out) {
  extern __shared__ char smem[];
  __bf16* xs   = (__bf16*)(smem);                              // x tile / attn-out
  __bf16* qkvs = (__bf16*)(smem + XS_BYTES);                   // q|k bf16
  __bf16* vT   = (__bf16*)(smem + XS_BYTES + QKV_BYTES);       // V transposed
  float*  scs  = (float*) (smem + XS_BYTES + QKV_BYTES + VT_BYTES);
  __bf16* ats  = (__bf16*)(smem + XS_BYTES + QKV_BYTES + VT_BYTES + SC_BYTES);
  float*  msk  = (float*) (smem + XS_BYTES + QKV_BYTES + VT_BYTES + SC_BYTES + AT_BYTES);

  const int tid  = threadIdx.x;
  const int lane = tid & 31;
  const int wave = tid >> 5;
  const int b    = blockIdx.x;          // window index
  const int wm   = b & (NMASK - 1);     // mask window = b % 64

  const float scale = 0.1767766952966369f;  // 32^-0.5

  // ---- Phase 0: kick off async LDS staging of this window's mask ----------
  // global_load_async_to_lds_b32: per-lane LDS[vdst] = MEM[vaddr], ASYNCcnt.
  // Issued first so the copy overlaps x staging and the QKV GEMM.
  {
    const float* msrc = mask + (size_t)wm * (NTOK * NTOK);
    const unsigned int lds0 = (unsigned int)(size_t)(void*)msk;
    for (int idx = tid; idx < NTOK * NTOK; idx += 256) {
      unsigned int ldst = lds0 + (unsigned int)(idx * 4);
      unsigned long long gsrc = (unsigned long long)(size_t)(msrc + idx);
      asm volatile("global_load_async_to_lds_b32 %0, %1, off"
                   :: "v"(ldst), "v"(gsrc) : "memory");
    }
  }

  // ---- Phase 1: stage x tile into LDS as bf16, zero-pad rows 49..63 -------
  {
    const float* xp = x + (size_t)b * NTOK * DIM;
    for (int idx = tid; idx < NP * DIM; idx += 256) {
      const int r = idx / DIM, c = idx - r * DIM;
      xs[r * XS_STRIDE + c] = (r < NTOK) ? (__bf16)xp[r * DIM + c] : (__bf16)0.0f;
    }
  }
  __syncthreads();

  // ---- Phase 2: QKV GEMM: (64x576) = xs(64x192) @ qkv_w^T + qkv_b ---------
  // q|k go to qkvs row-major; v is stored transposed into vT for fast PV.
  for (int nt = wave; nt < QKV_N / 16; nt += 8) {
    const int n0 = nt * 16;
    const float bias = qkv_b[n0 + (lane & 15)];
    v8f acc[4];
#pragma unroll
    for (int mt = 0; mt < 4; ++mt)
#pragma unroll
      for (int r = 0; r < 8; ++r) acc[mt][r] = bias;

#pragma unroll
    for (int kk = 0; kk < DIM / 32; ++kk) {
      const int k0 = kk * 32;
      v16bf bf = load_b_gweight(qkv_w, DIM, n0, k0, lane);
#pragma unroll
      for (int mt = 0; mt < 4; ++mt) {
        v16bf af = load_a16x32(xs, XS_STRIDE, mt * 16, k0, lane);
        acc[mt] = WMMA_BF16(af, bf, acc[mt]);
      }
    }
    if (n0 < 2 * DIM) {
      const float mul = (n0 < DIM) ? scale : 1.0f;  // q-block carries the scale
#pragma unroll
      for (int mt = 0; mt < 4; ++mt)
        store_d_bf16(qkvs, QKV_STRIDE, mt * 16, n0, lane, acc[mt], mul);
    } else {
#pragma unroll
      for (int mt = 0; mt < 4; ++mt)
        store_d_vT(vT, mt * 16, n0 - 2 * DIM, lane, acc[mt]);
    }
  }
  // Drain our async mask copies before the barrier that publishes LDS.
  asm volatile("s_wait_asynccnt 0x0" ::: "memory");
  __syncthreads();

  // ---- Phase 3: per-head attention -----------------------------------------
  for (int h = 0; h < HEADS; ++h) {
    // 3a: scores(64x64) = q_h(64x32) @ k_h^T ; K=32 == one bf16 WMMA step.
    for (int t = wave; t < 16; t += 8) {
      const int mi = t >> 2, ji = t & 3;
      v8f acc = {};
      v16bf qf = load_a16x32(qkvs + h * HD, QKV_STRIDE, mi * 16, 0, lane);
      v16bf kf = load_b_kcontig(qkvs + DIM + h * HD, QKV_STRIDE, ji * 16, 0, lane);
      acc = WMMA_BF16(qf, kf, acc);
      store_d_f32(scs, SC_STRIDE, mi * 16, ji * 16, lane, acc);
    }
    __syncthreads();

    // 3b: softmax, 4 lanes per row; bias from L2, mask from LDS; width-4
    //     shuffle reductions for row max/sum.
    {
      const int r = tid >> 2;   // 0..63
      const int q = tid & 3;
      if (r < NTOK) {
        const int*   ri   = rel_idx + r * NTOK;
        const float* mrow = msk + r * NTOK;
        float*       srow = scs + r * SC_STRIDE;
        float mx = -3.0e38f;
        for (int j = q; j < NTOK; j += 4) {
          float s = srow[j] + bias_table[ri[j] * HEADS + h] + mrow[j];
          srow[j] = s;
          mx = fmaxf(mx, s);
        }
        mx = fmaxf(mx, __shfl_xor(mx, 1, 4));
        mx = fmaxf(mx, __shfl_xor(mx, 2, 4));
        float sum = 0.0f;
        for (int j = q; j < NTOK; j += 4) {
          float e = __expf(srow[j] - mx);
          srow[j] = e;
          sum += e;
        }
        sum += __shfl_xor(sum, 1, 4);
        sum += __shfl_xor(sum, 2, 4);
        const float inv = 1.0f / sum;
        __bf16* arow = ats + r * AT_STRIDE;
        for (int j = q; j < NTOK; j += 4) arow[j] = (__bf16)(srow[j] * inv);
        for (int j = NTOK + q; j < NP; j += 4) arow[j] = (__bf16)0.0f;  // pad
      } else {
        __bf16* arow = ats + r * AT_STRIDE;
        for (int j = q; j < NP; j += 4) arow[j] = (__bf16)0.0f;
      }
    }
    __syncthreads();

    // 3c: out_h(64x32) = probs(64x64) @ v_h(64x32); 8 tiles == 1 per wave.
    //     V comes from vT so the B fragment is a contiguous 32B LDS load.
    {
      const int mi = wave >> 1, di = wave & 1;
      v8f acc = {};
#pragma unroll
      for (int jj = 0; jj < 2; ++jj) {
        v16bf af = load_a16x32(ats, AT_STRIDE, mi * 16, jj * 32, lane);
        v16bf vf = load_b_kcontig(vT + (h * HD + di * 16) * VT_STRIDE,
                                  VT_STRIDE, 0, jj * 32, lane);
        acc = WMMA_BF16(af, vf, acc);
      }
      // x tile is dead -> reuse xs as the attention output buffer (bf16).
      store_d_bf16(xs, XS_STRIDE, mi * 16, h * HD + di * 16, lane, acc, 1.0f);
    }
    __syncthreads();
  }

  // ---- Phase 4: projection: out(49x192) = attn_out(64x192) @ proj_w^T + b --
  float* outp = out + (size_t)b * NTOK * DIM;
  for (int t = wave; t < 48; t += 8) {
    const int nt = t >> 2, mt = t & 3;
    const int n0 = nt * 16;
    const float bias = proj_b[n0 + (lane & 15)];
    v8f acc;
#pragma unroll
    for (int r = 0; r < 8; ++r) acc[r] = bias;

#pragma unroll
    for (int kk = 0; kk < DIM / 32; ++kk) {
      const int k0 = kk * 32;
      v16bf bf = load_b_gweight(proj_w, DIM, n0, k0, lane);
      v16bf af = load_a16x32(xs, XS_STRIDE, mt * 16, k0, lane);
      acc = WMMA_BF16(af, bf, acc);
    }
    const int l  = lane & 15;
    const int hi = (lane >> 4) & 1;
#pragma unroll
    for (int r = 0; r < 8; ++r) {
      const int m = mt * 16 + hi * 8 + r;
      if (m < NTOK) outp[m * DIM + n0 + l] = acc[r];
    }
  }
}

// ---------------------------------------------------------------------------

extern "C" void kernel_launch(void* const* d_in, const int* in_sizes, int n_in,
                              void* d_out, int out_size, void* d_ws, size_t ws_size,
                              hipStream_t stream) {
  (void)in_sizes; (void)n_in; (void)out_size; (void)d_ws; (void)ws_size;
  const float* x          = (const float*)d_in[0];
  const float* mask       = (const float*)d_in[1];
  const float* qkv_w      = (const float*)d_in[2];
  const float* qkv_b      = (const float*)d_in[3];
  const float* proj_w     = (const float*)d_in[4];
  const float* proj_b     = (const float*)d_in[5];
  const float* bias_table = (const float*)d_in[6];
  const int*   rel_idx    = (const int*)d_in[7];

  hipLaunchKernelGGL(win_attn_fused_kernel, dim3(NWIN), dim3(256), SMEM_BYTES,
                     stream, x, mask, qkv_w, qkv_b, proj_w, proj_b, bias_table,
                     rel_idx, (float*)d_out);
}
